// SwinTransformerBlock_8710193676809
// MI455X (gfx1250) — matrix operand order repaired and checked by reading, not compiled, not run on hardware
//
#include <hip/hip_runtime.h>

typedef float  v8f   __attribute__((ext_vector_type(8)));
typedef __bf16 v16bf __attribute__((ext_vector_type(16)));

#define RES 56
#define WS  7
#define C   128
#define HID 512
#define NWIN 2048          // 32 * 8 * 8
#define NTOK 100352        // 32 * 56 * 56

__device__ inline v8f wmma_bf16(v16bf a, v16bf b, v8f c) {
  return __builtin_amdgcn_wmma_f32_16x16x32_bf16(false, a, false, b, (short)0, c,
                                                 false, false);
}

// A fragment: row-major [rows][rstride] bf16, tile at (row0, k0), K-dim = 32.
// lane 0-15: M=lane, K = k0+0..7 and k0+16..23; lane 16-31: M=lane-16, K = +8.
__device__ inline v16bf ldA(const __bf16* m, int rstride, int row0, int k0, int lane) {
  const __bf16* p = m + (row0 + (lane & 15)) * rstride + k0 + ((lane >> 4) << 3);
  v16bf a;
#pragma unroll
  for (int i = 0; i < 8; ++i) { a[i] = p[i]; a[8 + i] = p[16 + i]; }
  return a;
}

// B fragment from BT (= B transposed, [N][K] row-major, stride kstride).
// lane 0-15: N=n0+lane, K = k0+0..7 and k0+16..23; lane 16-31: N=n0+lane-16, K = +8.
__device__ inline v16bf ldB(const __bf16* bt, int kstride, int n0, int k0, int lane) {
  const __bf16* p = bt + (n0 + (lane & 15)) * kstride + k0 + ((lane >> 4) << 3);
  v16bf b;
#pragma unroll
  for (int i = 0; i < 8; ++i) { b[i] = p[i]; b[8 + i] = p[16 + i]; }
  return b;
}

// ---------------- weight conversion: fp32 [K][N] -> bf16 transposed [N][K] ---
__global__ __launch_bounds__(256) void convert_weights(
    const float* qkv_w, const float* proj_w, const float* w1, const float* w2,
    __bf16* qkvT, __bf16* projT, __bf16* w1T, __bf16* w2T) {
  int i = blockIdx.x * 256 + threadIdx.x;
  if (i < 384 * 128) { int o = i >> 7, k = i & 127; qkvT[i] = (__bf16)qkv_w[k * 384 + o]; }
  if (i < 128 * 128) { int o = i >> 7, k = i & 127; projT[i] = (__bf16)proj_w[k * 128 + o]; }
  if (i < 512 * 128) { int o = i >> 7, k = i & 127; w1T[i] = (__bf16)w1[k * 512 + o]; }
  if (i < 128 * 512) { int o = i >> 9, k = i & 511; w2T[i] = (__bf16)w2[k * 128 + o]; }
}

// ---------------- kernel 1: LN1 + window attention + proj + residual --------
__global__ __launch_bounds__(128) void attn_kernel(
    const float* __restrict__ x, const float* __restrict__ ln1g,
    const float* __restrict__ ln1b, const float* __restrict__ qkv_b,
    const float* __restrict__ proj_b, const __bf16* __restrict__ qkvT,
    const __bf16* __restrict__ projT, float* __restrict__ x1) {
  __shared__ __bf16 sA[64 * 128];      // LN'd window; later reused for O
  __shared__ __bf16 sQKV[64 * 384];    // Q|K|V per token
  __shared__ __bf16 sVt[128 * 64];     // V transposed: [h*32+d][token]
  __shared__ float  sS[4 * 64 * 64];   // per-head scores; P (bf16) overlays rows

  const int tid = threadIdx.x, lane = tid & 31, wave = tid >> 5;
  const int win = blockIdx.x;
  const int b = win >> 6, wy = (win >> 3) & 7, wx = win & 7;

  auto gaddr = [&](int t, int c) {
    int ty = t / 7, tx = t % 7;
    return (((b * RES + wy * WS + ty) * RES) + wx * WS + tx) * C + c;
  };

  // ---- Step 1: LayerNorm 1 -> sA (bf16), rows 49..63 zero --------------------
  for (int t = wave; t < 64; t += 4) {
    if (t < 49) {
      const float* px = x + gaddr(t, lane * 4);
      float v0 = px[0], v1 = px[1], v2 = px[2], v3 = px[3];
      float s = v0 + v1 + v2 + v3;
      float sq = v0 * v0 + v1 * v1 + v2 * v2 + v3 * v3;
#pragma unroll
      for (int off = 16; off; off >>= 1) {
        s  += __shfl_xor(s, off, 32);
        sq += __shfl_xor(sq, off, 32);
      }
      float mean = s * (1.f / 128.f);
      float inv = rsqrtf(sq * (1.f / 128.f) - mean * mean + 1e-5f);
      float vv[4] = {v0, v1, v2, v3};
#pragma unroll
      for (int i = 0; i < 4; ++i) {
        int c = lane * 4 + i;
        sA[t * 128 + c] = (__bf16)((vv[i] - mean) * inv * ln1g[c] + ln1b[c]);
      }
    } else {
#pragma unroll
      for (int i = 0; i < 4; ++i) sA[t * 128 + lane * 4 + i] = (__bf16)0.f;
    }
  }
  __syncthreads();

  // ---- Step 2: QKV = LN(x) @ qkv_w + b  (M=64, N=384, K=128) ----------------
  {
    const int mt = wave;
    v16bf afr[4];
#pragma unroll
    for (int ks = 0; ks < 4; ++ks) afr[ks] = ldA(sA, 128, mt * 16, ks * 32, lane);
    for (int nt = 0; nt < 24; ++nt) {
      v8f acc = {};
#pragma unroll
      for (int ks = 0; ks < 4; ++ks)
        acc = wmma_bf16(afr[ks], ldB(qkvT, 128, nt * 16, ks * 32, lane), acc);
      int n = nt * 16 + (lane & 15);
      float bias = qkv_b[n];
#pragma unroll
      for (int r = 0; r < 8; ++r) {
        int t = mt * 16 + ((lane >> 4) << 3) + r;
        float val = acc[r] + bias;
        sQKV[t * 384 + n] = (__bf16)val;
        if (n >= 256) sVt[(n - 256) * 64 + t] = (__bf16)val;   // V transposed
      }
    }
  }
  __syncthreads();

  // ---- Step 3: S = Q K^T * scale, wave-local softmax (one head per wave) ----
  {
    const int h = wave;
    const float scale = 0.1767766952966369f;  // 1/sqrt(32)
    float* Sh = sS + h * 64 * 64;
    for (int mt = 0; mt < 4; ++mt) {
      v16bf qa = ldA(sQKV, 384, mt * 16, h * 32, lane);
      for (int jt = 0; jt < 4; ++jt) {
        v8f s = {};
        s = wmma_bf16(qa, ldB(sQKV, 384, jt * 16, 128 + h * 32, lane), s);
#pragma unroll
        for (int r = 0; r < 8; ++r) {
          int m = ((lane >> 4) << 3) + r;
          Sh[(mt * 16 + m) * 64 + jt * 16 + (lane & 15)] = s[r] * scale;
        }
      }
    }
    // softmax over valid 49 keys; P (bf16, 64 cols) overlays the row in place
    for (int i = lane; i < 64; i += 32) {
      float* row = Sh + i * 64;
      float mx = -1e30f;
      for (int j = 0; j < 49; ++j) mx = fmaxf(mx, row[j]);
      float sum = 0.f;
      for (int j = 0; j < 49; ++j) sum += expf(row[j] - mx);
      float rinv = 1.f / sum;
      __bf16* prow = (__bf16*)row;   // ascending write never clobbers unread f32
      for (int j = 0; j < 64; ++j) {
        float p = (j < 49) ? expf(row[j] - mx) * rinv : 0.f;
        prow[j] = (__bf16)p;
      }
    }
  }

  // ---- Step 4: O = P @ V  (per head), into sA reused as O [64][128] ---------
  {
    const int h = wave;
    const __bf16* P = (const __bf16*)(sS + h * 64 * 64);  // row stride 128 bf16
    for (int mt = 0; mt < 4; ++mt) {
      v16bf pa0 = ldA(P, 128, mt * 16, 0, lane);
      v16bf pa1 = ldA(P, 128, mt * 16, 32, lane);
#pragma unroll
      for (int nt = 0; nt < 2; ++nt) {
        v8f acc = {};
        acc = wmma_bf16(pa0, ldB(sVt, 64, h * 32 + nt * 16, 0, lane), acc);
        acc = wmma_bf16(pa1, ldB(sVt, 64, h * 32 + nt * 16, 32, lane), acc);
#pragma unroll
        for (int r = 0; r < 8; ++r) {
          int m = ((lane >> 4) << 3) + r;
          sA[(mt * 16 + m) * 128 + h * 32 + nt * 16 + (lane & 15)] = (__bf16)acc[r];
        }
      }
    }
  }
  __syncthreads();

  // ---- Step 5: x1 = x + O @ proj_w + proj_b ---------------------------------
  {
    const int mt = wave;
    v16bf afr[4];
#pragma unroll
    for (int ks = 0; ks < 4; ++ks) afr[ks] = ldA(sA, 128, mt * 16, ks * 32, lane);
    for (int nt = 0; nt < 8; ++nt) {
      v8f acc = {};
#pragma unroll
      for (int ks = 0; ks < 4; ++ks)
        acc = wmma_bf16(afr[ks], ldB(projT, 128, nt * 16, ks * 32, lane), acc);
      int n = nt * 16 + (lane & 15);
      float bias = proj_b[n];
#pragma unroll
      for (int r = 0; r < 8; ++r) {
        int t = mt * 16 + ((lane >> 4) << 3) + r;
        if (t < 49) {
          int g = gaddr(t, n);
          x1[g] = x[g] + acc[r] + bias;
        }
      }
    }
  }
}

// ---------------- kernel 2: LN2 + MLP + residual ----------------------------
__global__ __launch_bounds__(128) void mlp_kernel(
    const float* __restrict__ x1, const float* __restrict__ ln2g,
    const float* __restrict__ ln2b, const float* __restrict__ b1,
    const float* __restrict__ b2, const __bf16* __restrict__ w1T,
    const __bf16* __restrict__ w2T, float* __restrict__ out) {
  __shared__ __bf16 sM[64 * 128];
  __shared__ __bf16 sH[64 * 512];

  const int tid = threadIdx.x, lane = tid & 31, wave = tid >> 5;
  const int base_tok = blockIdx.x * 64;

  // LN2 -> sM
  for (int t = wave; t < 64; t += 4) {
    const float* px = x1 + (base_tok + t) * 128 + lane * 4;
    float v0 = px[0], v1 = px[1], v2 = px[2], v3 = px[3];
    float s = v0 + v1 + v2 + v3;
    float sq = v0 * v0 + v1 * v1 + v2 * v2 + v3 * v3;
#pragma unroll
    for (int off = 16; off; off >>= 1) {
      s  += __shfl_xor(s, off, 32);
      sq += __shfl_xor(sq, off, 32);
    }
    float mean = s * (1.f / 128.f);
    float inv = rsqrtf(sq * (1.f / 128.f) - mean * mean + 1e-5f);
    float vv[4] = {v0, v1, v2, v3};
#pragma unroll
    for (int i = 0; i < 4; ++i) {
      int c = lane * 4 + i;
      sM[t * 128 + c] = (__bf16)((vv[i] - mean) * inv * ln2g[c] + ln2b[c]);
    }
  }
  __syncthreads();

  // H = gelu(M @ w1T + b1)   (M=64, N=512, K=128)
  {
    const int mt = wave;
    v16bf afr[4];
#pragma unroll
    for (int ks = 0; ks < 4; ++ks) afr[ks] = ldA(sM, 128, mt * 16, ks * 32, lane);
    for (int nt = 0; nt < 32; ++nt) {
      v8f acc = {};
#pragma unroll
      for (int ks = 0; ks < 4; ++ks)
        acc = wmma_bf16(afr[ks], ldB(w1T, 128, nt * 16, ks * 32, lane), acc);
      int n = nt * 16 + (lane & 15);
      float bias = b1[n];
#pragma unroll
      for (int r = 0; r < 8; ++r) {
        int m = ((lane >> 4) << 3) + r;
        float v = acc[r] + bias;
        v = 0.5f * v * (1.f + erff(v * 0.70710678118654752f));  // exact gelu
        sH[(mt * 16 + m) * 512 + n] = (__bf16)v;
      }
    }
  }
  __syncthreads();

  // out = x1 + H @ w2T + b2  (M=64, N=128, K=512)
  {
    const int mt = wave;
    for (int nt = 0; nt < 8; ++nt) {
      v8f acc = {};
      for (int ks = 0; ks < 16; ++ks)
        acc = wmma_bf16(ldA(sH, 512, mt * 16, ks * 32, lane),
                        ldB(w2T, 512, nt * 16, ks * 32, lane), acc);
      int n = nt * 16 + (lane & 15);
      float bias = b2[n];
#pragma unroll
      for (int r = 0; r < 8; ++r) {
        int row = base_tok + mt * 16 + ((lane >> 4) << 3) + r;
        int g = row * 128 + n;
        out[g] = x1[g] + acc[r] + bias;
      }
    }
  }
}

extern "C" void kernel_launch(void* const* d_in, const int* in_sizes, int n_in,
                              void* d_out, int out_size, void* d_ws, size_t ws_size,
                              hipStream_t stream) {
  const float* x      = (const float*)d_in[0];
  const float* ln1_g  = (const float*)d_in[1];
  const float* ln1_b  = (const float*)d_in[2];
  const float* qkv_w  = (const float*)d_in[3];
  const float* qkv_b  = (const float*)d_in[4];
  const float* proj_w = (const float*)d_in[5];
  const float* proj_b = (const float*)d_in[6];
  const float* ln2_g  = (const float*)d_in[7];
  const float* ln2_b  = (const float*)d_in[8];
  const float* mlp_w1 = (const float*)d_in[9];
  const float* mlp_b1 = (const float*)d_in[10];
  const float* mlp_w2 = (const float*)d_in[11];
  const float* mlp_b2 = (const float*)d_in[12];
  float* out = (float*)d_out;

  // workspace layout
  float*  x1    = (float*)d_ws;                 // 32*56*56*128 fp32
  __bf16* qkvT  = (__bf16*)(x1 + (size_t)NTOK * C);
  __bf16* projT = qkvT + 384 * 128;
  __bf16* w1T   = projT + 128 * 128;
  __bf16* w2T   = w1T + 512 * 128;

  convert_weights<<<256, 256, 0, stream>>>(qkv_w, proj_w, mlp_w1, mlp_w2,
                                           qkvT, projT, w1T, w2T);
  attn_kernel<<<NWIN, 128, 0, stream>>>(x, ln1_g, ln1_b, qkv_b, proj_b,
                                        qkvT, projT, x1);
  mlp_kernel<<<NTOK / 64, 128, 0, stream>>>(x1, ln2_g, ln2_b, mlp_b1, mlp_b2,
                                            w1T, w2T, out);
}
